// DFNPureModel_90417651515909
// MI455X (gfx1250) — matrix operand order repaired and checked by reading, not compiled, not run on hardware
//
#include <hip/hip_runtime.h>
#include <hip/hip_bf16.h>
#include <cstdint>
#include <cstddef>

typedef __bf16 bf16_t;
typedef __attribute__((ext_vector_type(16))) __bf16 v16bf;
typedef __attribute__((ext_vector_type(8)))  __bf16 v8bf;
typedef __attribute__((ext_vector_type(8)))  float  v8f;
typedef uint32_t u32x4 __attribute__((ext_vector_type(4)));
typedef int      i32x8 __attribute__((ext_vector_type(8)));
typedef int      i32x4 __attribute__((ext_vector_type(4)));

#if defined(__has_builtin)
#if __has_builtin(__builtin_amdgcn_tensor_load_to_lds) && __has_builtin(__builtin_amdgcn_s_wait_tensorcnt)
#define USE_TDM 1
#endif
#endif

__device__ __forceinline__ bf16_t f2bf(float f) {
  union { float f; uint32_t u; } c; c.f = f;
  uint32_t r = c.u + 0x7fffu + ((c.u >> 16) & 1u);   // round-to-nearest-even
  union { uint16_t s; bf16_t b; } o; o.s = (uint16_t)(r >> 16);
  return o.b;
}
__device__ __forceinline__ float bf2f(bf16_t b) {
  union { uint16_t s; bf16_t b; } i; i.b = b;
  union { uint32_t u; float f; } o; o.u = ((uint32_t)i.s) << 16;
  return o.f;
}
__device__ __forceinline__ float gelu_exact(float x) {
  return 0.5f * x * (1.0f + erff(x * 0.70710678118654752440f));
}

// ---------------------------------------------------------------------------
// Generic bf16 WMMA GEMM:  C[b] = act( A[b] (MxK) * Bt[b]^T (Bt is NxK) + bias )
// Block tile 128x128, BK=32, 256 threads = 8 waves in 2(M) x 4(N) layout,
// each wave computes a 64x32 region = 4x2 v_wmma_f32_16x16x32_bf16 tiles.
// Tile staging: Tensor Data Mover (tensor_load_to_lds) with double-buffered
// LDS; TDM pad feature reproduces the 40-half row stride (64B row + 16B pad),
// and hardware zero-fills beyond the tensor extents (handles K=514, M=10000).
// ---------------------------------------------------------------------------
#define BM 128
#define BN 128
#define BK 32
#define LDSW 40   // halves per LDS row; 80 bytes = 5*16 -> rows stay 16B aligned
#define TILE_HALVES (BM * LDSW)

#ifdef USE_TDM
// Issue one 2D TDM tile load: rows x BK bf16 elements from row-major src
// (leading dim ldK elements) into LDS at lds_addr with 16B pad per 64B row.
// rem_k / rem_rows are the remaining tensor extents measured from the tile
// origin; TDM zero-fills reads beyond them.
__device__ __forceinline__ void tdm_load_tile(uint32_t lds_addr, const bf16_t* src,
                                              int rem_k, int rem_rows, int ldK) {
  const uint64_t ga = (uint64_t)(uintptr_t)src;
  u32x4 g0;
  g0[0] = 1u;                                   // count=1 (valid), load, no gather
  g0[1] = lds_addr;                             // LDS byte address
  g0[2] = (uint32_t)ga;                         // global_addr[31:0]
  g0[3] = (uint32_t)((ga >> 32) & 0x1FFFFFFu)   // global_addr[56:32]
        | (2u << 30);                           // type=2 ("image")
  const uint32_t dim0 = (uint32_t)(rem_k   > 0 ? rem_k   : 0);
  const uint32_t dim1 = (uint32_t)(rem_rows > 0 ? rem_rows : 0);
  const uint64_t str0 = (uint64_t)(uint32_t)ldK;
  i32x8 g1;
  g1[0] = (int)((1u << 16)      // data_size = 2 bytes
              | (1u << 20)      // pad_enable
              | (3u << 22)      // pad_interval: 16 DWORDs (one 64B tile row)
              | (3u << 25));    // pad_amount: 4 DWORDs (16B)
  g1[1] = (int)((dim0 & 0xFFFFu) << 16);                       // tensor_dim0 lo
  g1[2] = (int)((dim0 >> 16) | ((dim1 & 0xFFFFu) << 16));      // dim0 hi | dim1 lo
  g1[3] = (int)((dim1 >> 16) | ((uint32_t)BK << 16));          // dim1 hi | tile_dim0
  g1[4] = (int)(uint32_t)BM;                                   // tile_dim1 (tile_dim2=0)
  g1[5] = (int)(uint32_t)str0;                                 // tensor_dim0_stride lo
  g1[6] = (int)(uint32_t)(str0 >> 32);                         // stride hi (dim1_stride=0)
  g1[7] = 0;
  const i32x4 z4 = {0, 0, 0, 0};
  const i32x8 z8 = {0, 0, 0, 0, 0, 0, 0, 0};
  // clang-23 / therock-10.0 6-arg form: (g0, g1, g2, g3, g4, cpol)
  __builtin_amdgcn_tensor_load_to_lds(g0, g1, z4, z4, z8, 0);
}
#endif

__global__ __launch_bounds__(256)
void wmma_gemm_bf16(const bf16_t* __restrict__ A,
                    const bf16_t* __restrict__ Bt,
                    const float*  __restrict__ bias,
                    void* __restrict__ Cout,
                    int M, int N, int K,
                    long long strideA, long long strideB, long long strideC,
                    int actGelu, int outBf16)
{
  __shared__ __align__(16) bf16_t As[2 * TILE_HALVES];
  __shared__ __align__(16) bf16_t Bs[2 * TILE_HALVES];

  const int b = blockIdx.z;
  A  += (long long)b * strideA;
  Bt += (long long)b * strideB;

  const int tile_m = blockIdx.y * BM;
  const int tile_n = blockIdx.x * BN;

  const int tid  = threadIdx.x;
  const int wave = tid >> 5;
  const int lane = tid & 31;
  const int wm   = wave >> 2;   // 0..1  (64 rows each)
  const int wn   = wave & 3;    // 0..3  (32 cols each)
  const int lrow = lane & 15;
  const int lhi  = lane >> 4;   // 0/1

  const v8f vzero = {0.f,0.f,0.f,0.f,0.f,0.f,0.f,0.f};
  v8f acc[4][2];
#pragma unroll
  for (int i = 0; i < 4; ++i)
#pragma unroll
    for (int j = 0; j < 2; ++j) acc[i][j] = vzero;

  const int nt = (K + BK - 1) / BK;

#ifdef USE_TDM
  // ---- double-buffered TDM pipeline: wave 0 issues DMA for tile t+2 once
  // ---- all waves finish WMMA on tile t; it overlaps WMMA on t+1. TENSORcnt: 2 per tile (A + B).
  if (wave == 0) {
    tdm_load_tile((uint32_t)(uintptr_t)&As[0], A + (long long)tile_m * K,
                  K, M - tile_m, K);
    tdm_load_tile((uint32_t)(uintptr_t)&Bs[0], Bt + (long long)tile_n * K,
                  K, N - tile_n, K);
    if (nt > 1) {
      tdm_load_tile((uint32_t)(uintptr_t)&As[TILE_HALVES],
                    A + (long long)tile_m * K + BK, K - BK, M - tile_m, K);
      tdm_load_tile((uint32_t)(uintptr_t)&Bs[TILE_HALVES],
                    Bt + (long long)tile_n * K + BK, K - BK, N - tile_n, K);
    }
  }
#endif

  for (int it = 0; it < nt; ++it) {
    const int buf = it & 1;
#ifdef USE_TDM
    if (wave == 0) {
      if (it + 1 < nt) __builtin_amdgcn_s_wait_tensorcnt(2);  // tile `it` done
      else             __builtin_amdgcn_s_wait_tensorcnt(0);  // last tile done
    }
    __syncthreads();   // publish tile `it` to all waves
#else
    // ---- fallback staging: clamped addresses + select, no exec branching ----
    {
      const int k0 = it * BK;
      const int r  = tid >> 1;
      const int kp = (tid & 1) * 16;
      {
        const int grow  = tile_m + r;
        const int rsafe = grow < M ? grow : (M - 1);
        const bf16_t* src = A + (long long)rsafe * K;
        bf16_t* dst = &As[buf * TILE_HALVES + r * LDSW + kp];
#pragma unroll
        for (int e = 0; e < 16; ++e) {
          const int gk = k0 + kp + e;
          const int ks = gk < K ? gk : (K - 1);
          const bf16_t v = src[ks];
          dst[e] = (grow < M && gk < K) ? v : f2bf(0.0f);
        }
      }
      {
        const int gcol  = tile_n + r;
        const int csafe = gcol < N ? gcol : (N - 1);
        const bf16_t* src = Bt + (long long)csafe * K;
        bf16_t* dst = &Bs[buf * TILE_HALVES + r * LDSW + kp];
#pragma unroll
        for (int e = 0; e < 16; ++e) {
          const int gk = k0 + kp + e;
          const int ks = gk < K ? gk : (K - 1);
          const bf16_t v = src[ks];
          dst[e] = (gcol < N && gk < K) ? v : f2bf(0.0f);
        }
      }
    }
    __syncthreads();
#endif

    // ---- fragments per documented wave32 layouts, then WMMA ----
    const bf16_t* Ab = &As[buf * TILE_HALVES];
    const bf16_t* Bb = &Bs[buf * TILE_HALVES];
    v16bf afr[4];
#pragma unroll
    for (int i = 0; i < 4; ++i) {
      const int row = wm * 64 + i * 16 + lrow;
      const v8bf* pa = (const v8bf*)&Ab[row * LDSW];   // 16B aligned
      v8bf lo = pa[lhi];        // K = 8*lhi .. 8*lhi+7
      v8bf hi = pa[lhi + 2];    // K = 16+8*lhi ..
      afr[i] = __builtin_shufflevector(lo, hi, 0,1,2,3,4,5,6,7,8,9,10,11,12,13,14,15);
    }
    v16bf bfr[2];
#pragma unroll
    for (int j = 0; j < 2; ++j) {
      const int col = wn * 32 + j * 16 + lrow;
      const v8bf* pb = (const v8bf*)&Bb[col * LDSW];
      v8bf lo = pb[lhi];
      v8bf hi = pb[lhi + 2];
      bfr[j] = __builtin_shufflevector(lo, hi, 0,1,2,3,4,5,6,7,8,9,10,11,12,13,14,15);
    }
#pragma unroll
    for (int i = 0; i < 4; ++i)
#pragma unroll
      for (int j = 0; j < 2; ++j)
        acc[i][j] = __builtin_amdgcn_wmma_f32_16x16x32_bf16(
            false, afr[i], false, bfr[j], (short)0, acc[i][j], false, false);

    __syncthreads();   // all waves done reading buf before it is refilled

#ifdef USE_TDM
    if (wave == 0 && it + 2 < nt) {
      const int k2 = (it + 2) * BK;
      tdm_load_tile((uint32_t)(uintptr_t)&As[buf * TILE_HALVES],
                    A + (long long)tile_m * K + k2, K - k2, M - tile_m, K);
      tdm_load_tile((uint32_t)(uintptr_t)&Bs[buf * TILE_HALVES],
                    Bt + (long long)tile_n * K + k2, K - k2, N - tile_n, K);
    }
#endif
  }

  // ---- epilogue: bias + optional exact GELU, f32 or bf16 store ----
  float*  Cf = (float*)Cout;
  bf16_t* Cb = (bf16_t*)Cout;
  const long long coff = (long long)b * strideC;
#pragma unroll
  for (int i = 0; i < 4; ++i) {
#pragma unroll
    for (int j = 0; j < 2; ++j) {
      const int ncol = tile_n + wn * 32 + j * 16 + lrow;
      if (ncol >= N) continue;
      const float bv = bias ? bias[ncol] : 0.0f;
#pragma unroll
      for (int e = 0; e < 8; ++e) {
        const int mrow = tile_m + wm * 64 + i * 16 + e + (lhi ? 8 : 0);
        if (mrow >= M) continue;
        float v = acc[i][j][e] + bv;
        if (actGelu) v = gelu_exact(v);
        if (outBf16) Cb[coff + (long long)mrow * N + ncol] = f2bf(v);
        else         Cf[coff + (long long)mrow * N + ncol] = v;
      }
    }
  }
}

// ---------------------------------------------------------------------------
// Elementwise / helper kernels
// ---------------------------------------------------------------------------
__global__ void k_f32_to_bf16(const float* __restrict__ in, bf16_t* __restrict__ out,
                              long long n) {
  long long i = (long long)blockIdx.x * blockDim.x + threadIdx.x;
  const long long stride = (long long)gridDim.x * blockDim.x;
  for (; i < n; i += stride) out[i] = f2bf(in[i]);
}

// in (K,N) f32 -> out (N,K) bf16, batched via blockIdx.z
__global__ void k_transpose_bf16(const float* __restrict__ in, bf16_t* __restrict__ out,
                                 int K, int N, long long sIn, long long sOut) {
  const long long b = blockIdx.z;
  in  += b * sIn;
  out += b * sOut;
  const int n = blockIdx.x * 16 + (threadIdx.x & 15);
  const int k = blockIdx.y * 16 + (threadIdx.x >> 4);
  if (n < N && k < K) out[(long long)n * K + k] = f2bf(in[(long long)k * N + n]);
}

// coords = df @ Wc + bc -> field[:,0:2]; one block per row
__global__ void k_coords(const bf16_t* __restrict__ df, const float* __restrict__ Wc,
                         const float* __restrict__ bc, bf16_t* __restrict__ field) {
  __shared__ float s0[256], s1[256];
  const long long rrow = blockIdx.x;
  const bf16_t* row = df + rrow * 512;
  float a0 = 0.f, a1 = 0.f;
  for (int k = threadIdx.x; k < 512; k += 256) {
    const float v = bf2f(row[k]);
    a0 += v * Wc[k * 2 + 0];
    a1 += v * Wc[k * 2 + 1];
  }
  s0[threadIdx.x] = a0; s1[threadIdx.x] = a1;
  __syncthreads();
  for (int o = 128; o > 0; o >>= 1) {
    if (threadIdx.x < (unsigned)o) {
      s0[threadIdx.x] += s0[threadIdx.x + o];
      s1[threadIdx.x] += s1[threadIdx.x + o];
    }
    __syncthreads();
  }
  if (threadIdx.x == 0) {
    field[rrow * 514 + 0] = f2bf(s0[0] + bc[0]);
    field[rrow * 514 + 1] = f2bf(s1[0] + bc[1]);
  }
}

// field[:,2:514] = df
__global__ void k_field_copy(const bf16_t* __restrict__ df, bf16_t* __restrict__ field,
                             long long M) {
  long long i = (long long)blockIdx.x * blockDim.x + threadIdx.x;
  const long long total = M * 512;
  const long long stride = (long long)gridDim.x * blockDim.x;
  for (; i < total; i += stride) {
    const long long r = i >> 9;
    const int k = (int)(i & 511);
    field[r * 514 + 2 + k] = df[i];
  }
}

// imp[row] = sum(ef[row,:]^2)   (squared norm; monotone for top-k)
__global__ void k_rownorm2(const float* __restrict__ ef, float* __restrict__ imp, int cols) {
  __shared__ float s[256];
  const long long r = blockIdx.x;
  const float* row = ef + r * (long long)cols;
  float a = 0.f;
  for (int k = threadIdx.x; k < cols; k += 256) { const float v = row[k]; a += v * v; }
  s[threadIdx.x] = a;
  __syncthreads();
  for (int o = 128; o > 0; o >>= 1) {
    if (threadIdx.x < (unsigned)o) s[threadIdx.x] += s[threadIdx.x + o];
    __syncthreads();
  }
  if (threadIdx.x == 0) imp[r] = s[0];
}

// top-Ksel indices per batch row of S values (set semantics; order arbitrary)
__global__ void k_topk(const float* __restrict__ imp, int* __restrict__ topidx,
                       int S, int Ksel) {
  __shared__ unsigned sv[4096];
  __shared__ int cnt;
  __shared__ int wpos;
  const int b = blockIdx.x;
  const float* row = imp + (long long)b * S;
  for (int i = threadIdx.x; i < S; i += 256) sv[i] = __float_as_uint(row[i]);
  __syncthreads();
  unsigned thr = 0u;
  for (int bit = 31; bit >= 0; --bit) {
    const unsigned cand = thr | (1u << bit);
    if (threadIdx.x == 0) cnt = 0;
    __syncthreads();
    int local = 0;
    for (int i = threadIdx.x; i < S; i += 256) local += (sv[i] >= cand) ? 1 : 0;
    atomicAdd(&cnt, local);
    __syncthreads();
    if (cnt >= Ksel) thr = cand;
    __syncthreads();
  }
  if (threadIdx.x == 0) wpos = 0;
  __syncthreads();
  for (int i = threadIdx.x; i < S; i += 256) {
    if (sv[i] > thr) {
      const int p = atomicAdd(&wpos, 1);
      topidx[(long long)b * Ksel + p] = i;
    }
  }
  __syncthreads();
  for (int i = threadIdx.x; i < S; i += 256) {
    if (sv[i] == thr) {
      const int p = atomicAdd(&wpos, 1);
      if (p < Ksel) topidx[(long long)b * Ksel + p] = i;
    }
  }
}

// gather selected rows: positions (f32) and states (bf16)
__global__ void k_gather(const float* __restrict__ ef, const int* __restrict__ topidx,
                         float* __restrict__ positions, bf16_t* __restrict__ states,
                         int S, int cols, int Ksel, int ED) {
  const int bn = blockIdx.x;                 // b*Ksel + n
  const int b = bn / Ksel;
  const int idx = topidx[bn];
  const float* src = ef + ((long long)b * S + idx) * cols;
  bf16_t* dst = states + (long long)bn * ED;
  for (int d = threadIdx.x; d < ED; d += 256) dst[d] = f2bf(src[2 + d]);
  if (threadIdx.x == 0) {
    positions[bn * 2 + 0] = src[0];
    positions[bn * 2 + 1] = src[1];
  }
}

// normalized Gaussian attention weights -> bf16; one block per (b,g)
__global__ void k_attn(const float* __restrict__ positions, bf16_t* __restrict__ attnb,
                       int G, int Nent, int gs) {
  __shared__ float s[256];
  __shared__ float sinv;
  const int bg = blockIdx.x;
  const int b = bg / G, g = bg % G;
  const int gi = g / gs, gj = g % gs;
  const float step = 2.0f / (float)(gs - 1);
  const float gx = -1.0f + step * (float)gi;
  const float gy = -1.0f + step * (float)gj;
  const float* pos = positions + (long long)b * Nent * 2;
  float wv[2];
  float lsum = 0.f;
#pragma unroll
  for (int t = 0; t < 2; ++t) {
    const int n = threadIdx.x + t * 256;
    const float dx = gx - pos[n * 2 + 0];
    const float dy = gy - pos[n * 2 + 1];
    const float w = __expf(-(dx * dx + dy * dy) * 10.0f);   // exp(-sq/0.1)
    wv[t] = w;
    lsum += w;
  }
  s[threadIdx.x] = lsum;
  __syncthreads();
  for (int o = 128; o > 0; o >>= 1) {
    if (threadIdx.x < (unsigned)o) s[threadIdx.x] += s[threadIdx.x + o];
    __syncthreads();
  }
  if (threadIdx.x == 0) sinv = 1.0f / (s[0] + 1e-8f);
  __syncthreads();
  bf16_t* outp = attnb + (long long)bg * Nent;
#pragma unroll
  for (int t = 0; t < 2; ++t) outp[threadIdx.x + t * 256] = f2bf(wv[t] * sinv);
}

// ---------------------------------------------------------------------------
static void launch_gemm(const bf16_t* A, const bf16_t* Bt, const float* bias, void* C,
                        int M, int N, int K,
                        long long sA, long long sB, long long sC, int batch,
                        int act, int outBf16, hipStream_t stream) {
  dim3 grid((N + BN - 1) / BN, (M + BM - 1) / BM, batch);
  wmma_gemm_bf16<<<grid, 256, 0, stream>>>(A, Bt, bias, C, M, N, K, sA, sB, sC, act, outBf16);
}

extern "C" void kernel_launch(void* const* d_in, const int* in_sizes, int n_in,
                              void* d_out, int out_size, void* d_ws, size_t ws_size,
                              hipStream_t stream) {
  (void)in_sizes; (void)n_in; (void)out_size; (void)ws_size;
  const float* x   = (const float*)d_in[0];
  const float* W1  = (const float*)d_in[1];
  const float* b1  = (const float*)d_in[2];
  const float* W2  = (const float*)d_in[3];
  const float* b2  = (const float*)d_in[4];
  const float* Wc  = (const float*)d_in[5];
  const float* bc  = (const float*)d_in[6];
  const float* We1 = (const float*)d_in[7];
  const float* be1 = (const float*)d_in[8];
  const float* We2 = (const float*)d_in[9];
  const float* be2 = (const float*)d_in[10];
  const float* Wo1 = (const float*)d_in[11];
  const float* bo1 = (const float*)d_in[12];
  const float* Wo2 = (const float*)d_in[13];
  const float* bo2 = (const float*)d_in[14];
  float* out = (float*)d_out;

  const int Mrows = 8 * 4096;   // 32768 token rows
  const int IN = 512;           // INPUT_DIM
  const int FD = 512;           // FIELD_DIM
  const int ED = 1024;          // ENTITY_DIM
  const int CK = FD + 2;        // 514 (field_input K)
  const int EC = ED + 3;        // 1027 (ef cols)
  const int S  = 4096;
  const int Bb = 8;
  const int Ksel = 512;         // NUM_ENTITIES
  const int G  = 10000, GS = 100;

  size_t off = 0;
  auto alloc = [&](size_t bytes) -> void* {
    void* p = (char*)d_ws + off;
    off += (bytes + 255) & ~(size_t)255;
    return p;
  };
  bf16_t* xb    = (bf16_t*)alloc((size_t)Mrows * IN * 2);
  bf16_t* W1t   = (bf16_t*)alloc((size_t)FD * IN * 2);
  bf16_t* W2t   = (bf16_t*)alloc((size_t)FD * FD * 2);
  bf16_t* We1t  = (bf16_t*)alloc((size_t)FD * CK * 2);
  bf16_t* We2t  = (bf16_t*)alloc((size_t)EC * FD * 2);
  bf16_t* Wo1t  = (bf16_t*)alloc((size_t)(2 * ED) * ED * 2);
  bf16_t* Wo2t  = (bf16_t*)alloc((size_t)IN * (2 * ED) * 2);
  bf16_t* h1    = (bf16_t*)alloc((size_t)Mrows * FD * 2);
  bf16_t* df    = (bf16_t*)alloc((size_t)Mrows * FD * 2);
  bf16_t* field = (bf16_t*)alloc((size_t)Mrows * CK * 2);
  float*  ef    = (float*)alloc((size_t)Mrows * EC * 4);
  float*  imp   = (float*)alloc((size_t)Mrows * 4);
  int*    tidx  = (int*)alloc((size_t)Bb * Ksel * 4);
  float*  pos   = (float*)alloc((size_t)Bb * Ksel * 2 * 4);
  bf16_t* states= (bf16_t*)alloc((size_t)Bb * Ksel * ED * 2);
  float*  proj  = (float*)alloc((size_t)Bb * Ksel * IN * 4);
  bf16_t* projT = (bf16_t*)alloc((size_t)Bb * IN * Ksel * 2);
  // overlays (lifetimes disjoint with their hosts):
  bf16_t* he    = h1;              // h1 dead after GEMM2
  bf16_t* ho    = xb;              // xb dead after GEMM1; 16MB <= 32MB
  bf16_t* attnb = (bf16_t*)ef;     // ef dead after gather; 82MB <= 134MB

  // 1) x -> bf16
  k_f32_to_bf16<<<4096, 256, 0, stream>>>(x, xb, (long long)Mrows * IN);
  // 2) weights -> bf16, transposed (N x K)
  k_transpose_bf16<<<dim3(IN / 16 + 1, FD / 16 + 1, 1), 256, 0, stream>>>(W1, W1t, IN, FD, 0, 0);
  k_transpose_bf16<<<dim3(FD / 16 + 1, FD / 16 + 1, 1), 256, 0, stream>>>(W2, W2t, FD, FD, 0, 0);
  k_transpose_bf16<<<dim3(FD / 16 + 1, CK / 16 + 1, 1), 256, 0, stream>>>(We1, We1t, CK, FD, 0, 0);
  k_transpose_bf16<<<dim3(EC / 16 + 1, FD / 16 + 1, 1), 256, 0, stream>>>(We2, We2t, FD, EC, 0, 0);
  k_transpose_bf16<<<dim3((2 * ED) / 16 + 1, ED / 16 + 1, 1), 256, 0, stream>>>(Wo1, Wo1t, ED, 2 * ED, 0, 0);
  k_transpose_bf16<<<dim3(IN / 16 + 1, (2 * ED) / 16 + 1, 1), 256, 0, stream>>>(Wo2, Wo2t, 2 * ED, IN, 0, 0);

  // 3) h1 = gelu(x @ W1 + b1)
  launch_gemm(xb, W1t, b1, h1, Mrows, FD, IN, 0, 0, 0, 1, /*gelu*/1, /*bf16*/1, stream);
  // 4) df = gelu(h1 @ W2 + b2)
  launch_gemm(h1, W2t, b2, df, Mrows, FD, FD, 0, 0, 0, 1, 1, 1, stream);
  // 5) field = [coords | df]
  k_coords<<<Mrows, 256, 0, stream>>>(df, Wc, bc, field);
  k_field_copy<<<4096, 256, 0, stream>>>(df, field, Mrows);
  // 6) he = gelu(field @ We1 + be1)
  launch_gemm(field, We1t, be1, he, Mrows, FD, CK, 0, 0, 0, 1, 1, 1, stream);
  // 7) ef = he @ We2 + be2   (f32 out)
  launch_gemm(he, We2t, be2, ef, Mrows, EC, FD, 0, 0, 0, 1, 0, 0, stream);
  // 8) importance + top-k + gather
  k_rownorm2<<<Mrows, 256, 0, stream>>>(ef, imp, EC);
  k_topk<<<Bb, 256, 0, stream>>>(imp, tidx, S, Ksel);
  k_gather<<<Bb * Ksel, 256, 0, stream>>>(ef, tidx, pos, states, S, EC, Ksel, ED);
  // 9) ho = gelu(states @ Wo1 + bo1)
  launch_gemm(states, Wo1t, bo1, ho, Bb * Ksel, 2 * ED, ED, 0, 0, 0, 1, 1, 1, stream);
  // 10) proj = ho @ Wo2 + bo2  (f32 out)
  launch_gemm(ho, Wo2t, bo2, proj, Bb * Ksel, IN, 2 * ED, 0, 0, 0, 1, 0, 0, stream);
  // 11) projT (per batch: D x Nent, bf16)
  k_transpose_bf16<<<dim3(IN / 16 + 1, Ksel / 16 + 1, Bb), 256, 0, stream>>>(
      proj, projT, Ksel, IN, (long long)Ksel * IN, (long long)IN * Ksel);
  // 12) attention weights (bf16)
  k_attn<<<Bb * G, 256, 0, stream>>>(pos, attnb, G, Ksel, GS);
  // 13) out = attn @ proj  (batched WMMA GEMM, f32 out)
  launch_gemm(attnb, projT, nullptr, out, G, IN, Ksel,
              (long long)G * Ksel, (long long)IN * Ksel, (long long)G * IN,
              Bb, 0, 0, stream);
}
